// Co_Guiding_GAT_65377992180060
// MI455X (gfx1250) — matrix operand order repaired and checked by reading, not compiled, not run on hardware
//
#include <hip/hip_runtime.h>
#include <hip/hip_bf16.h>
#include <stdint.h>

// ---------------------------------------------------------------------------
// Co-guiding GAT forward for MI455X (gfx1250).
//  * all matmuls via v_wmma_f32_16x16x32_f16 (f16 in, f32 accumulate)
//  * flash-style attention (no [B,H,N,N] tensor)
//  * GEMM A-tile: ping-pong global_load_async_to_lds_b128 (ASYNCcnt),
//    issued post-barrier so the copy overlaps the current tile's WMMAs
//  * attention K-tile: ping-pong TDM tensor_load_to_lds (TENSORcnt)
// ---------------------------------------------------------------------------

typedef __attribute__((ext_vector_type(16))) _Float16 v16h;
typedef __attribute__((ext_vector_type(8)))  _Float16 v8h;
typedef __attribute__((ext_vector_type(8)))  float    v8f;
typedef __attribute__((ext_vector_type(4)))  int      i4;
typedef __attribute__((ext_vector_type(8)))  int      i8;

#define DEV __device__ __forceinline__

constexpr int Bb = 4;     // batch
constexpr int Nn = 1024;  // nodes
constexpr int Dd = 256;   // model dim
constexpr int Hh = 8;     // heads
constexpr int DK = 32;    // head dim
constexpr int Ll = 2;     // layers
constexpr size_t HN = (size_t)Bb * Nn * Dd;
constexpr size_t WSZ = (size_t)Ll * 4 * Dd * Dd;

// LDS byte address of a __shared__ object (async / TDM destinations).
typedef __attribute__((address_space(3))) void lds_void;
DEV uint32_t lds_addr_of(const void* p) {
  return (uint32_t)(uintptr_t)(lds_void*)(void*)p;
}

// Wave-wide async global->LDS 16B/lane copy (tracked by ASYNCcnt).
DEV void async_load_b128(uint32_t lds_addr, uint32_t voff, const void* sbase) {
  asm volatile("global_load_async_to_lds_b128 %0, %1, %2"
               :: "v"(lds_addr), "v"(voff), "s"(sbase)
               : "memory");
}
DEV void wait_async0() {
  asm volatile("s_wait_asynccnt 0x0" ::: "memory");
}

// TDM: 2D tile (tile_dim0 x tile_dim1 elems, 2B each) global -> LDS.
DEV void tdm_load_tile_f16(uint32_t lds_addr, uint64_t gaddr,
                           uint32_t tile_d0, uint32_t tile_d1,
                           uint32_t tensor_d0, uint32_t tensor_d1,
                           uint32_t stride0) {
  i4 g0;
  g0[0] = 1;                                  // count=1, user descriptor
  g0[1] = (int)lds_addr;                      // LDS byte address
  g0[2] = (int)(uint32_t)gaddr;               // global addr [31:0]
  g0[3] = (int)(((uint32_t)(gaddr >> 32) & 0x01ffffffu) | (2u << 30)); // type=2
  i8 g1;
  g1[0] = (int)(1u << 16);                    // data_size = 1 (2 bytes)
  g1[1] = (int)((tensor_d0 & 0xffffu) << 16); // tensor_dim0 lo16 @ bits 63:48
  g1[2] = (int)((tensor_d0 >> 16) | ((tensor_d1 & 0xffffu) << 16));
  g1[3] = (int)(((tensor_d1 >> 16) & 0xffffu) | (tile_d0 << 16));
  g1[4] = (int)(tile_d1 & 0xffffu);           // tile_dim1 (tile_dim2 = 0)
  g1[5] = (int)stride0;                       // tensor_dim0_stride [31:0]
  g1[6] = 0;
  g1[7] = 0;
  asm volatile("tensor_load_to_lds %0, %1" :: "s"(g0), "s"(g1) : "memory");
}
DEV void wait_tensor0() {
  asm volatile("s_wait_tensorcnt 0x0" ::: "memory");
}

// ---------------------------------------------------------------------------
// WMMA fragment loaders (documented gfx1250 VGPR layouts).
// A (16x32): lane L: M=L&15, h=L>>4; elems 0..7 = K(8h..), 8..15 = K(16+8h..)
// B (32x16): lane L: N=L&15, h=L>>4; elems 0..7 = K(8h..), 8..15 = K(16+8h..)  (tile stored Bt[n][k])
// C (16x16): lane L: N=L&15; VGPR j: M=j+8*(L>>4)
// ---------------------------------------------------------------------------
DEV v16h load_a_frag(const _Float16* base, int ld) {
  const int lane = threadIdx.x & 31;
  const int r = lane & 15, hh = lane >> 4;
  const _Float16* p = base + r * ld;
  v8h lo = *(const v8h*)(p + 8 * hh);
  v8h hi = *(const v8h*)(p + 16 + 8 * hh);
  v16h out;
#pragma unroll
  for (int i = 0; i < 8; ++i) { out[i] = lo[i]; out[i + 8] = hi[i]; }
  return out;
}

DEV v16h load_b_frag(const _Float16* base, int ld) {
  const int lane = threadIdx.x & 31;
  const int n = lane & 15, hh = lane >> 4;
  const _Float16* p = base + n * ld + 8 * hh;
  v8h lo = *(const v8h*)(p);
  v8h hi = *(const v8h*)(p + 16);
  v16h out;
#pragma unroll
  for (int i = 0; i < 8; ++i) { out[i] = lo[i]; out[i + 8] = hi[i]; }
  return out;
}

DEV v8f wmma_f16(v16h a, v16h b, v8f c) {
  return __builtin_amdgcn_wmma_f32_16x16x32_f16(false, a, false, b,
                                                (short)0, c, false, false);
}

// ---------------------------------------------------------------------------
// GEMM: out = X(f16)[MxK] @ W(f16)[KxN] + bias (+ residual).
// 64x64 block tile, 8 waves x (16x32).  Double-buffered software pipeline:
//   loop top : commit W regs -> Bt[cur], wait async A[cur], barrier
//   post-bar : issue async A[nxt], fetch W[nxt] to regs (overlaps WMMAs)
// ---------------------------------------------------------------------------
__global__ __launch_bounds__(256) void gemm_f16_kernel(
    const _Float16* __restrict__ X, const _Float16* __restrict__ W,
    const float* __restrict__ bias, const float* __restrict__ residual,
    _Float16* __restrict__ out16, float* __restrict__ out32,
    int M, int N, int K) {
  __shared__ _Float16 As[2][64 * 32];
  __shared__ _Float16 Bt[2][64 * 32];

  const int t = threadIdx.x;
  const int lane = t & 31;
  const int wave = t >> 5;
  const int m0 = blockIdx.y * 64;
  const int n0 = blockIdx.x * 64;
  const int wm = wave & 3;
  const int wn = wave >> 2;

  v8f acc0 = {}, acc1 = {};

  const int r_a = t >> 2, c_a = (t & 3) * 8;  // A stage: 64 rows x 4 vec8
  const int k_b = t >> 3, n_b = (t & 7) * 8;  // B stage: 32 rows x 8 vec8

  const _Float16* Xrow = X + (size_t)m0 * K;
  const uint32_t ldsA[2] = {lds_addr_of(&As[0][r_a * 32 + c_a]),
                            lds_addr_of(&As[1][r_a * 32 + c_a])};
  const _Float16* wp = &W[(size_t)k_b * N + n0 + n_b];
  const size_t wstep = (size_t)32 * N;
  uint32_t aoff = (uint32_t)((r_a * K + c_a) * 2);

  // Prologue: tile 0 in flight.
  async_load_b128(ldsA[0], aoff, Xrow);
  v8h wv = *(const v8h*)wp;

  const int KT = K / 32;
  for (int kt = 0; kt < KT; ++kt) {
    const int cur = kt & 1;
    _Float16* bt = &Bt[cur][0];
    // Commit the pre-fetched W tile to LDS (transposed).
#pragma unroll
    for (int e = 0; e < 8; ++e) bt[(n_b + e) * 32 + k_b] = wv[e];
    wait_async0();          // A[cur] landed (issued one iteration ago)
    __syncthreads();        // publish As[cur] / Bt[cur]

    const bool more = (kt + 1 < KT);
    if (more) {             // overlap next tile's traffic with the WMMAs
      aoff += 64;
      async_load_b128(ldsA[cur ^ 1], aoff, Xrow);
      wp += wstep;
      wv = *(const v8h*)wp;
      if (kt + 2 < KT) __builtin_prefetch(wp + wstep, 0, 3);
    }

    const _Float16* as = &As[cur][0];
    v16h a  = load_a_frag(as + wm * 16 * 32, 32);
    v16h b0 = load_b_frag(bt + (wn * 32) * 32, 32);
    v16h b1 = load_b_frag(bt + (wn * 32 + 16) * 32, 32);
    acc0 = wmma_f16(a, b0, acc0);
    acc1 = wmma_f16(a, b1, acc1);
  }

  const int hh = lane >> 4, nn = lane & 15;
#pragma unroll
  for (int f = 0; f < 2; ++f) {
    v8f acc = f ? acc1 : acc0;
    const int col = n0 + wn * 32 + f * 16 + nn;
    const float bv = bias ? bias[col] : 0.f;
#pragma unroll
    for (int j = 0; j < 8; ++j) {
      const int row = m0 + wm * 16 + j + 8 * hh;
      float v = acc[j] + bv;
      if (residual) v += residual[(size_t)row * N + col];
      if (out16) out16[(size_t)row * N + col] = (_Float16)v;
      if (out32) out32[(size_t)row * N + col] = v;
    }
  }
}

// ---------------------------------------------------------------------------
// Fused adjacency-masked flash attention.
// grid=(B*H, N/128); block=256 (8 waves); wave owns 16 q-rows.
// K tile: ping-pong TDM (wave 0).  V tile: register-pipelined transpose.
// One barrier per k-block; next tile's TDM issued post-barrier.
// ---------------------------------------------------------------------------
__global__ __launch_bounds__(256) void attn_kernel(
    const _Float16* __restrict__ Q, const _Float16* __restrict__ Km,
    const _Float16* __restrict__ V, const int* __restrict__ adj,
    _Float16* __restrict__ O) {
  __shared__ _Float16 Ks[2][32 * 32];   // K tile row-major: [kcol][d]
  __shared__ _Float16 Vt[2][32 * 32];   // V tile transposed: [d][kcol]
  __shared__ _Float16 Ps[8 * 16 * 32];  // per-wave P tile (16 x 32 f16)

  const int t = threadIdx.x, lane = t & 31, wave = t >> 5;
  const int bh = blockIdx.x;
  const int b = bh / Hh, h = bh % Hh;
  const int q0 = blockIdx.y * 128 + wave * 16;
  const int hh = lane >> 4, nn = lane & 15;

  // Preload the wave's constant Q fragment from global memory.
  v16h aq;
  {
    const _Float16* qp = Q + ((size_t)(b * Nn + q0 + nn) * Dd) + h * DK;
    v8h lo = *(const v8h*)(qp + 8 * hh);
    v8h hi = *(const v8h*)(qp + 16 + 8 * hh);
#pragma unroll
    for (int i = 0; i < 8; ++i) { aq[i] = lo[i]; aq[i + 8] = hi[i]; }
  }

  v8f acc0 = {}, acc1 = {};
  float m_r[8], l_r[8];
#pragma unroll
  for (int j = 0; j < 8; ++j) { m_r[j] = -3.0e38f; l_r[j] = 0.f; }

  const int t2 = t & 127;
  const int sr = t2 >> 2, sc = (t2 & 3) * 8;   // 32 rows x 4 vec8
  const float scale = 0.17677669529663687f;     // 1/sqrt(32)
  const uint32_t ldsKs[2] = {lds_addr_of(&Ks[0][0]), lds_addr_of(&Ks[1][0])};
  const uint64_t kbase =
      (uint64_t)(uintptr_t)(Km + ((size_t)b * Nn * Dd) + h * DK);

  // Prologue: K tile 0 via TDM; V tile 0 into registers.
  if (wave == 0) {
    tdm_load_tile_f16(ldsKs[0], kbase, 32, 32, Dd, (uint32_t)(Bb * Nn), Dd);
  }
  const _Float16* vp = V + ((size_t)(b * Nn + sr) * Dd) + h * DK + sc;
  v8h vv = (t < 128) ? *(const v8h*)vp : v8h{};

  const int KTiles = Nn / 32;
  for (int kt = 0; kt < KTiles; ++kt) {
    const int cur = kt & 1;
    const int kb = kt * 32;
    // Commit pre-fetched V tile (transposed) to LDS.
    if (t < 128) {
#pragma unroll
      for (int e = 0; e < 8; ++e) Vt[cur][(sc + e) * 32 + sr] = vv[e];
    }
    if (wave == 0) wait_tensor0();  // Ks[cur] landed
    __syncthreads();

    const bool more = (kt + 1 < KTiles);
    if (more) {  // overlap next K/V tile traffic with this block's compute
      if (wave == 0) {
        tdm_load_tile_f16(ldsKs[cur ^ 1], kbase + (size_t)(kb + 32) * Dd * 2,
                          32, 32, Dd, (uint32_t)(Bb * Nn), Dd);
      }
      if (t < 128) {
        vp += (size_t)32 * Dd;
        vv = *(const v8h*)vp;
      }
    }

    // Scores S[16q x 32k]: two WMMAs over K-dim = d_k = 32.
    v8f s0 = {}, s1 = {};
    s0 = wmma_f16(aq, load_b_frag(&Ks[cur][0], 32), s0);
    s1 = wmma_f16(aq, load_b_frag(&Ks[cur][16 * 32], 32), s1);

    // Scale + adjacency mask (exact -1e9 like the reference).
#pragma unroll
    for (int j = 0; j < 8; ++j) {
      const int row = q0 + j + 8 * hh;
      const int* ap = adj + ((size_t)b * Nn + row) * Nn + kb;
      float v0 = s0[j] * scale; if (ap[nn] == 0)      v0 = -1e9f;
      float v1 = s1[j] * scale; if (ap[16 + nn] == 0) v1 = -1e9f;
      s0[j] = v0; s1[j] = v1;
    }

    // Row max across the 16-lane half (C rows live in lane halves).
    float mb[8];
#pragma unroll
    for (int j = 0; j < 8; ++j) mb[j] = fmaxf(s0[j], s1[j]);
#pragma unroll
    for (int mask = 1; mask < 16; mask <<= 1)
#pragma unroll
      for (int j = 0; j < 8; ++j)
        mb[j] = fmaxf(mb[j], __shfl_xor(mb[j], mask, 32));

    // Online softmax update.
    float pl[8];
#pragma unroll
    for (int j = 0; j < 8; ++j) {
      const float mn = fmaxf(m_r[j], mb[j]);
      const float rs = __expf(m_r[j] - mn);
      m_r[j] = mn;
      l_r[j] *= rs;
      acc0[j] *= rs; acc1[j] *= rs;
      const float p0 = __expf(s0[j] - mn);
      const float p1 = __expf(s1[j] - mn);
      s0[j] = p0; s1[j] = p1;
      pl[j] = p0 + p1;
    }
#pragma unroll
    for (int mask = 1; mask < 16; mask <<= 1)
#pragma unroll
      for (int j = 0; j < 8; ++j) pl[j] += __shfl_xor(pl[j], mask, 32);
#pragma unroll
    for (int j = 0; j < 8; ++j) l_r[j] += pl[j];

    // Spill P (C layout) to per-wave LDS, reload as an A fragment.
    _Float16* pw = &Ps[wave * 512];
#pragma unroll
    for (int j = 0; j < 8; ++j) {
      pw[(j + 8 * hh) * 32 + nn]      = (_Float16)s0[j];
      pw[(j + 8 * hh) * 32 + 16 + nn] = (_Float16)s1[j];
    }
    v16h ap2 = load_a_frag(pw, 32);
    acc0 = wmma_f16(ap2, load_b_frag(&Vt[cur][0], 32), acc0);
    acc1 = wmma_f16(ap2, load_b_frag(&Vt[cur][16 * 32], 32), acc1);
  }

  // O = acc / l, merged back to [B,N,D] with head offset.
#pragma unroll
  for (int f = 0; f < 2; ++f) {
    v8f acc = f ? acc1 : acc0;
#pragma unroll
    for (int j = 0; j < 8; ++j) {
      const int row = q0 + j + 8 * hh;
      const float o = acc[j] / l_r[j];
      O[((size_t)(b * Nn + row) * Dd) + h * DK + f * 16 + nn] = (_Float16)o;
    }
  }
}

// ---------------------------------------------------------------------------
// LayerNorm over D=256: one wave per row.
// ---------------------------------------------------------------------------
__global__ __launch_bounds__(256) void ln_kernel(
    const float* __restrict__ x, const float* __restrict__ g,
    const float* __restrict__ be, float* __restrict__ y, int rows) {
  const int wave = threadIdx.x >> 5, lane = threadIdx.x & 31;
  const int row = blockIdx.x * 8 + wave;
  if (row >= rows) return;
  const float* xr = x + (size_t)row * Dd;
  float4 v0 = *(const float4*)(xr + lane * 8);
  float4 v1 = *(const float4*)(xr + lane * 8 + 4);
  float s  = v0.x + v0.y + v0.z + v0.w + v1.x + v1.y + v1.z + v1.w;
  float ss = v0.x * v0.x + v0.y * v0.y + v0.z * v0.z + v0.w * v0.w +
             v1.x * v1.x + v1.y * v1.y + v1.z * v1.z + v1.w * v1.w;
#pragma unroll
  for (int mask = 1; mask < 32; mask <<= 1) {
    s  += __shfl_xor(s, mask, 32);
    ss += __shfl_xor(ss, mask, 32);
  }
  const float mean = s * (1.f / Dd);
  const float var  = ss * (1.f / Dd) - mean * mean;
  const float inv  = rsqrtf(var + 1e-5f);
  const float xv[8] = {v0.x, v0.y, v0.z, v0.w, v1.x, v1.y, v1.z, v1.w};
#pragma unroll
  for (int e = 0; e < 8; ++e) {
    const int c = lane * 8 + e;
    y[(size_t)row * Dd + c] = (xv[e] - mean) * inv * g[c] + be[c];
  }
}

// ---------------------------------------------------------------------------
// Elementwise helpers.
// ---------------------------------------------------------------------------
__global__ void combine_kernel(const float* __restrict__ a,
                               const float* __restrict__ b,
                               const float* __restrict__ orig,
                               float* __restrict__ h32,
                               _Float16* __restrict__ h16,
                               float* __restrict__ fout, int n) {
  const int i = blockIdx.x * 256 + threadIdx.x;
  if (i >= n) return;
  float v = a[i] + b[i];
  v = v > 0.f ? v : 0.f;
  h32[i] = v;
  h16[i] = (_Float16)v;
  if (fout) fout[i] = orig[i] + v;
}

__global__ void f32_to_f16_kernel(const float* __restrict__ x,
                                  _Float16* __restrict__ y, int n) {
  const int i = blockIdx.x * 256 + threadIdx.x;
  if (i < n) y[i] = (_Float16)x[i];
}

__global__ void init_h_kernel(const float* __restrict__ x,
                              float* __restrict__ h32,
                              _Float16* __restrict__ h16, int n) {
  const int i = blockIdx.x * 256 + threadIdx.x;
  if (i < n) { const float v = x[i]; h32[i] = v; h16[i] = (_Float16)v; }
}

// ---------------------------------------------------------------------------
// Host orchestration.
// ---------------------------------------------------------------------------
extern "C" void kernel_launch(void* const* d_in, const int* in_sizes, int n_in,
                              void* d_out, int out_size, void* d_ws,
                              size_t ws_size, hipStream_t stream) {
  const float* h_a_in = (const float*)d_in[0];
  const float* h_b_in = (const float*)d_in[1];
  const int* adjsel[4] = {(const int*)d_in[2], (const int*)d_in[3],
                          (const int*)d_in[4], (const int*)d_in[5]};
  const float* Wq = (const float*)d_in[6];
  const float* bq = (const float*)d_in[7];
  const float* Wk = (const float*)d_in[8];
  const float* bk = (const float*)d_in[9];
  const float* Wv = (const float*)d_in[10];
  const float* bv = (const float*)d_in[11];
  const float* Wo = (const float*)d_in[12];
  const float* bo = (const float*)d_in[13];
  const float* ln_g = (const float*)d_in[14];
  const float* ln_b = (const float*)d_in[15];
  float* outA = (float*)d_out;
  float* outB = outA + HN;

  char* w = (char*)d_ws;
  auto alloc = [&](size_t bytes) -> void* {
    void* p = w;
    w += (bytes + 255) & ~(size_t)255;
    return p;
  };
  _Float16* Wq16 = (_Float16*)alloc(WSZ * 2);
  _Float16* Wk16 = (_Float16*)alloc(WSZ * 2);
  _Float16* Wv16 = (_Float16*)alloc(WSZ * 2);
  _Float16* Wo16 = (_Float16*)alloc(WSZ * 2);
  float*    hA32 = (float*)alloc(HN * 4);
  float*    hB32 = (float*)alloc(HN * 4);
  _Float16* hA16 = (_Float16*)alloc(HN * 2);
  _Float16* hB16 = (_Float16*)alloc(HN * 2);
  _Float16* Q16  = (_Float16*)alloc(HN * 2);
  _Float16* K16  = (_Float16*)alloc(HN * 2);
  _Float16* V16  = (_Float16*)alloc(HN * 2);
  _Float16* AT16 = (_Float16*)alloc(HN * 2);
  float*    P32  = (float*)alloc(HN * 4);
  float*    LN32[4];
  for (int j = 0; j < 4; ++j) LN32[j] = (float*)alloc(HN * 4);

  const int wblk = (int)(WSZ / 256);
  f32_to_f16_kernel<<<wblk, 256, 0, stream>>>(Wq, Wq16, (int)WSZ);
  f32_to_f16_kernel<<<wblk, 256, 0, stream>>>(Wk, Wk16, (int)WSZ);
  f32_to_f16_kernel<<<wblk, 256, 0, stream>>>(Wv, Wv16, (int)WSZ);
  f32_to_f16_kernel<<<wblk, 256, 0, stream>>>(Wo, Wo16, (int)WSZ);
  const int hblk = (int)(HN / 256);
  init_h_kernel<<<hblk, 256, 0, stream>>>(h_a_in, hA32, hA16, (int)HN);
  init_h_kernel<<<hblk, 256, 0, stream>>>(h_b_in, hB32, hB16, (int)HN);

  auto gemm = [&](const _Float16* X, const _Float16* Wt, const float* bias,
                  const float* res, _Float16* o16, float* o32) {
    dim3 grid(Dd / 64, (Bb * Nn) / 64);
    gemm_f16_kernel<<<grid, 256, 0, stream>>>(X, Wt, bias, res, o16, o32,
                                              Bb * Nn, Dd, Dd);
  };

  for (int i = 0; i < Ll; ++i) {
    const _Float16* qs16[4]  = {hA16, hB16, hB16, hA16};
    const float*    qs32[4]  = {hA32, hB32, hB32, hA32};
    const _Float16* kvs16[4] = {hA16, hB16, hA16, hB16};
    for (int j = 0; j < 4; ++j) {
      const size_t wo = (size_t)(i * 4 + j) * Dd * Dd;
      const size_t bo_ = (size_t)(i * 4 + j) * Dd;
      gemm(qs16[j],  Wq16 + wo, bq + bo_, nullptr, Q16, nullptr);
      gemm(kvs16[j], Wk16 + wo, bk + bo_, nullptr, K16, nullptr);
      gemm(kvs16[j], Wv16 + wo, bv + bo_, nullptr, V16, nullptr);
      attn_kernel<<<dim3(Bb * Hh, Nn / 128), 256, 0, stream>>>(
          Q16, K16, V16, adjsel[j], AT16);
      gemm(AT16, Wo16 + wo, bo + bo_, qs32[j], nullptr, P32);
      ln_kernel<<<(Bb * Nn) / 8, 256, 0, stream>>>(
          P32, ln_g + bo_, ln_b + bo_, LN32[j], Bb * Nn);
    }
    const bool last = (i == Ll - 1);
    combine_kernel<<<hblk, 256, 0, stream>>>(
        LN32[0], LN32[3], last ? h_a_in : nullptr, hA32, hA16,
        last ? outA : nullptr, (int)HN);
    combine_kernel<<<hblk, 256, 0, stream>>>(
        LN32[1], LN32[2], last ? h_b_in : nullptr, hB32, hB16,
        last ? outB : nullptr, (int)HN);
  }
}
